// LocalMatching_12086037971338
// MI455X (gfx1250) — matrix operand order repaired and checked by reading, not compiled, not run on hardware
//
#include <hip/hip_runtime.h>

#define QL   30
#define PARA 100
#define EMB  300
#define NPAR 1024
#define NIMG 2048
#define C1O  50
#define C2O  40
#define FCIN 5520

typedef __attribute__((ext_vector_type(16))) __bf16 v16bf;
typedef __attribute__((ext_vector_type(8)))  float  v8f;
typedef __attribute__((ext_vector_type(4)))  unsigned v4u;
typedef __attribute__((ext_vector_type(8)))  int      v8i;
typedef __attribute__((ext_vector_type(4)))  int      v4i;

union BFV { v16bf v; unsigned short u[16]; };

#if __has_builtin(__builtin_amdgcn_tensor_load_to_lds) && __has_builtin(__builtin_amdgcn_s_wait_tensorcnt)
#define HAVE_TDM 1
#else
#define HAVE_TDM 0
#endif

__device__ __forceinline__ unsigned short f2bf(float f) {
  union { float f; unsigned u; } x; x.f = f;
  unsigned u = x.u;
  return (unsigned short)((u + 0x7FFFu + ((u >> 16) & 1u)) >> 16);
}
// monotonic float<->uint map so LDS atomicMax(u32) == float max (handles negatives)
__device__ __forceinline__ unsigned fToKey(float f) {
  union { float f; unsigned u; } x; x.f = f;
  return (x.u & 0x80000000u) ? ~x.u : (x.u | 0x80000000u);
}
__device__ __forceinline__ float keyToF(unsigned k) {
  union { unsigned u; float f; } x;
  x.u = (k & 0x80000000u) ? (k & 0x7FFFFFFFu) : ~k;
  return x.f;
}
__device__ __forceinline__ float waveSum(float v) {
  for (int o = 16; o > 0; o >>= 1) v += __shfl_down(v, o);
  return __shfl(v, 0);
}
__device__ __forceinline__ float waveMax(float v) {
  for (int o = 16; o > 0; o >>= 1) v = fmaxf(v, __shfl_down(v, o));
  return __shfl(v, 0);
}

#if HAVE_TDM
// 1-D TDM copy: n8 x 8-byte elements, global -> LDS byte offset lds_off.
// D# per CDNA5 ISA ch.8: group0 = {count=1 | lds_addr | 57b global_addr | type=2},
// group1 = {data_size=3(8B) | tensor_dim0=n8 | tensor_dim1=1 | tile_dim0=n8 | stride0=n8}.
__device__ __forceinline__ void tdm_copy_to_lds(unsigned lds_off, const void* g, unsigned n8) {
  unsigned long long ga = (unsigned long long)(uintptr_t)g;
  v4u g0 = { 1u, lds_off, (unsigned)ga,
             ((unsigned)((ga >> 32) & 0x01FFFFFFu)) | 0x80000000u };
  v8i g1 = { (int)0x00030000u,                       // data_size=3 (8B)
             (int)((n8 & 0xFFFFu) << 16),            // tensor_dim0[15:0]
             (int)(((n8 >> 16) & 0xFFFFu) | (1u << 16)), // tensor_dim0[31:16] | tensor_dim1=1
             (int)((n8 & 0xFFFFu) << 16),            // tile_dim0
             0,                                      // tile_dim1/2 unused
             (int)n8,                                // tensor_dim0_stride[31:0]
             0, 0 };
  v4i z = { 0, 0, 0, 0 };
#if defined(__clang_major__) && (__clang_major__ >= 23)
  v8i z8 = { 0, 0, 0, 0, 0, 0, 0, 0 };
  __builtin_amdgcn_tensor_load_to_lds(g0, g1, z, z, z8, 0);
#else
  __builtin_amdgcn_tensor_load_to_lds(g0, g1, z, z, 0);
#endif
}
#endif

// ---------------- K0: prep qn (bf16, padded 32x320), tx, conv weight B-tiles ----
__global__ void k0_prep(const int* quer, const float* emb, const float* w_t,
                        const float* b_t, const float* c1w, const float* c2w,
                        unsigned short* qnb, float* tx,
                        unsigned short* B1, unsigned short* B2) {
  int tid = threadIdx.x, lane = tid & 31, wid = tid >> 5;
  if (blockIdx.x == 0) {
    for (int q = wid; q < QL; q += 8) {
      int tok = quer[q];
      float ssq = 0.f, sw = 0.f;
      for (int t = 0; t < 10; t++) {
        int c = lane + 32 * t;
        float v = (c < EMB && tok != 1) ? emb[(size_t)tok * EMB + c] : 0.f;
        ssq += v * v;
        sw  += (c < EMB) ? v * w_t[c] : 0.f;
      }
      ssq = waveSum(ssq); sw = waveSum(sw);
      float denom = fmaxf(sqrtf(ssq), 1e-12f);
      for (int t = 0; t < 10; t++) {
        int c = lane + 32 * t;
        if (c < 320) {
          float v = (c < EMB && tok != 1) ? emb[(size_t)tok * EMB + c] / denom : 0.f;
          qnb[q * 320 + c] = f2bf(v);
        }
      }
      if (lane == 0) tx[q] = sw + b_t[0];
    }
    for (int i = tid; i < 2 * 320; i += blockDim.x) qnb[QL * 320 + 640 - 640 + i + 0] = 0, qnb[30 * 320 + i] = 0;
  }
  int stride = gridDim.x * blockDim.x;
  int gid = blockIdx.x * blockDim.x + tid;
  for (int i = gid; i < 32 * 64; i += stride) {            // conv1 weights -> B1[k=32][n=64]
    int k = i >> 6, n = i & 63;
    B1[i] = (k < 27 && n < C1O) ? f2bf(c1w[n * 27 + k]) : (unsigned short)0;
  }
  for (int i = gid; i < 480 * 48; i += stride) {           // conv2 weights -> B2[k=480][n=48]
    int k = i / 48, n = i % 48;
    B2[i] = (k < 450 && n < C2O) ? f2bf(c2w[n * 450 + k]) : (unsigned short)0;
  }
}

// ---------------- K1: per-paragraph gather/normalize + WMMA M_cos + channels ----
__global__ void k1_match(const int* quer, const int* doc, const float* emb,
                         const float* w_t, const float* b_t,
                         const unsigned short* qnb, const float* tx,
                         unsigned short* xbuf) {
  extern __shared__ char smem[];
  unsigned short* dn   = (unsigned short*)smem;            // 112*320 bf16 (71680 B)
  float*          tyl  = (float*)(smem + 71680);           // 112
  float*          txl  = (float*)(smem + 72128);           // 32
  int*            qtok = (int*)(smem + 72256);             // 32
  int*            dtok = (int*)(smem + 72384);             // 128
  unsigned short* qnbl = (unsigned short*)(smem + 72896);  // 32*320 bf16 (20480 B)
  int tid = threadIdx.x, lane = tid & 31, wid = tid >> 5;
  int p = blockIdx.x;

#if HAVE_TDM
  if (tid == 0) tdm_copy_to_lds(72896u, qnb, 20480u / 8u);
#else
  for (int i = tid; i < 32 * 320; i += blockDim.x) qnbl[i] = qnb[i];
#endif

  if (tid < QL)  { qtok[tid] = quer[tid]; txl[tid] = tx[tid]; }
  if (tid < PARA) dtok[tid] = doc[p * PARA + tid];
  __syncthreads();

  for (int j = wid; j < 112; j += 8) {
    if (j < PARA) {
      int tok = dtok[j];
      float ssq = 0.f, sw = 0.f;
      for (int t = 0; t < 10; t++) {
        int c = lane + 32 * t;
        float v = (c < EMB && tok != 1) ? emb[(size_t)tok * EMB + c] : 0.f;
        ssq += v * v;
        sw  += (c < EMB) ? v * w_t[c] : 0.f;
      }
      ssq = waveSum(ssq); sw = waveSum(sw);
      float denom = fmaxf(sqrtf(ssq), 1e-12f);
      for (int t = 0; t < 10; t++) {
        int c = lane + 32 * t;
        if (c < 320) {
          float v = (c < EMB && tok != 1) ? emb[(size_t)tok * EMB + c] / denom : 0.f;
          dn[j * 320 + c] = f2bf(v);
        }
      }
      if (lane == 0) tyl[j] = sw + b_t[0];
    } else {
      for (int t = 0; t < 10; t++) { int c = lane + 32 * t; if (c < 320) dn[j * 320 + c] = 0; }
    }
  }
#if HAVE_TDM
  if (tid == 0) __builtin_amdgcn_s_wait_tensorcnt(0);
#endif
  __syncthreads();

  size_t img1 = (size_t)(NPAR + p);
  int mrow = lane & 15, half = lane >> 4;
  for (int t = wid; t < 14; t += 8) {                      // 2 m-tiles x 7 n-tiles
    int tm = t / 7, tn = t % 7;
    v8f acc = {};
    for (int kk = 0; kk < 10; kk++) {                      // K = 320
      BFV a, b;
      #pragma unroll
      for (int i = 0; i < 16; i++) {
        int ka = kk * 32 + i + 8 * half + ((i >= 8) ? 8 : 0);
        a.u[i] = qnbl[(tm * 16 + mrow) * 320 + ka];
        int kb = kk * 32 + i + 8 * half + ((i >= 8) ? 8 : 0);
        b.u[i] = dn[(tn * 16 + mrow) * 320 + kb];
      }
      acc = __builtin_amdgcn_wmma_f32_16x16x32_bf16(false, a.v, false, b.v,
                                                    (short)0, acc, false, false);
    }
    #pragma unroll
    for (int r = 0; r < 8; r++) {
      int q  = tm * 16 + r + 8 * half;
      int pc = tn * 16 + (lane & 15);
      if (q < QL && pc < PARA)
        xbuf[(img1 * 3 + 2) * 3000 + q * 100 + pc] = f2bf(1.0f - acc[r]);
    }
  }

  size_t img0 = (size_t)p;
  for (int idx = tid; idx < 3000; idx += blockDim.x) {
    int q = idx / 100, pc = idx % 100;
    unsigned short ta = f2bf(txl[q]);
    unsigned short tb = f2bf(tyl[pc]);
    unsigned short xr = (qtok[q] == dtok[pc]) ? f2bf(1.0f) : (unsigned short)0;
    xbuf[(img0 * 3 + 0) * 3000 + idx] = ta;
    xbuf[(img0 * 3 + 1) * 3000 + idx] = tb;
    xbuf[(img0 * 3 + 2) * 3000 + idx] = xr;
    xbuf[(img1 * 3 + 0) * 3000 + idx] = ta;
    xbuf[(img1 * 3 + 1) * 3000 + idx] = tb;
  }
}

// -------- K2: fused conv1+pool -> conv2+pool -> fc -> log_softmax (per image) ---
// LDS byte offsets (dynamic): sS 0, sB1 18048, sB2 22144, p1 68224, p2 205424,
//                             p2f 227504, sfc 249584; total 249840 B (<320KB/WGP)
__global__ void k2_conv(const unsigned short* xbuf,
                        const unsigned short* B1g, const unsigned short* B2g,
                        const float* c1b, const float* c2b,
                        const float* fcw, const float* fcb, float* out) {
  extern __shared__ char smem[];
  unsigned short* sS  = (unsigned short*)(smem + 0);       // 9024
  unsigned short* sB1 = (unsigned short*)(smem + 18048);   // 2048
  unsigned short* sB2 = (unsigned short*)(smem + 22144);   // 23040
  unsigned*       p1  = (unsigned*)(smem + 68224);         // 34300 (50x14x49 keys)
  unsigned*       p2  = (unsigned*)(smem + 205424);        // 5520  (40x6x23 keys)
  float*          p2f = (float*)(smem + 227504);           // 5520
  float*          sfc = (float*)(smem + 249584);           // 64
  int tid = threadIdx.x, lane = tid & 31, wid = tid >> 5;
  int n = blockIdx.x;
  const unsigned NEGKEY = fToKey(-3.0e38f);

#if HAVE_TDM
  if (tid == 0) {                                          // async DMA: global -> LDS
    tdm_copy_to_lds(0u,     xbuf + (size_t)n * 9000, 18000u / 8u);
    tdm_copy_to_lds(18048u, B1g,                     4096u  / 8u);
    tdm_copy_to_lds(22144u, B2g,                     46080u / 8u);
  }
#else
  for (int i = tid; i < 9000;  i += blockDim.x) sS[i]  = xbuf[(size_t)n * 9000 + i];
  for (int i = tid; i < 2048;  i += blockDim.x) sB1[i] = B1g[i];
  for (int i = tid; i < 23040; i += blockDim.x) sB2[i] = B2g[i];
#endif
  for (int i = tid; i < 34300; i += blockDim.x) p1[i] = NEGKEY;
  for (int i = tid; i < 5520;  i += blockDim.x) p2[i] = NEGKEY;
#if HAVE_TDM
  if (tid == 0) __builtin_amdgcn_s_wait_tensorcnt(0);
#endif
  __syncthreads();

  int mrow = lane & 15, half = lane >> 4;

  // conv1: M=2744 (172 tiles), 4 n-tiles, K=27->32 (single WMMA per tile)
  for (int t = wid; t < 172 * 4; t += 8) {
    int tm = t >> 2, tn = t & 3;
    int m = tm * 16 + mrow;
    int oh = m / 98, ow = m % 98;
    BFV a, b;
    #pragma unroll
    for (int i = 0; i < 16; i++) {
      int ka = i + 8 * half + ((i >= 8) ? 8 : 0);
      unsigned short av = 0;
      if (m < 2744 && ka < 27) {
        int c = ka / 9, r9 = ka % 9;
        av = sS[c * 3000 + (oh + r9 / 3) * 100 + (ow + r9 % 3)];
      }
      a.u[i] = av;
      b.u[i] = sB1[(i + 8 * half + ((i >= 8) ? 8 : 0)) * 64 + tn * 16 + mrow];
    }
    v8f acc = {};
    acc = __builtin_amdgcn_wmma_f32_16x16x32_bf16(false, a.v, false, b.v,
                                                  (short)0, acc, false, false);
    #pragma unroll
    for (int r = 0; r < 8; r++) {
      int mm = tm * 16 + r + 8 * half, nn = tn * 16 + (lane & 15);
      if (mm < 2744 && nn < C1O) {
        int oh2 = mm / 98, ow2 = mm % 98;
        atomicMax(&p1[nn * 686 + (oh2 >> 1) * 49 + (ow2 >> 1)], fToKey(acc[r] + c1b[nn]));
      }
    }
  }
  __syncthreads();

  // conv2: M=564 (36 tiles), 3 n-tiles, K=450->480 (15 WMMA per tile)
  for (int t = wid; t < 36 * 3; t += 8) {
    int tm = t / 3, tn = t % 3;
    int m = tm * 16 + mrow;
    int oh = m / 47, ow = m % 47;
    v8f acc = {};
    for (int kk = 0; kk < 15; kk++) {
      BFV a, b;
      #pragma unroll
      for (int i = 0; i < 16; i++) {
        int ka = kk * 32 + i + 8 * half + ((i >= 8) ? 8 : 0);
        unsigned short av = 0;
        if (m < 564 && ka < 450) {
          int c = ka / 9, r9 = ka % 9;
          float f = keyToF(p1[c * 686 + (oh + r9 / 3) * 49 + (ow + r9 % 3)]);
          av = f2bf(fmaxf(f, 0.f));                        // relu(pool1)
        }
        a.u[i] = av;
        b.u[i] = sB2[(kk * 32 + i + 8 * half + ((i >= 8) ? 8 : 0)) * 48 + tn * 16 + mrow];
      }
      acc = __builtin_amdgcn_wmma_f32_16x16x32_bf16(false, a.v, false, b.v,
                                                    (short)0, acc, false, false);
    }
    #pragma unroll
    for (int r = 0; r < 8; r++) {
      int mm = tm * 16 + r + 8 * half, nn = tn * 16 + (lane & 15);
      if (mm < 564 && nn < C2O) {
        int oh2 = mm / 47, ow2 = mm % 47;
        if (ow2 < 46)
          atomicMax(&p2[nn * 138 + (oh2 >> 1) * 23 + (ow2 >> 1)], fToKey(acc[r] + c2b[nn]));
      }
    }
  }
  __syncthreads();

  for (int i = tid; i < FCIN; i += blockDim.x) p2f[i] = fmaxf(keyToF(p2[i]), 0.f);
  __syncthreads();

  for (int j = wid; j < C1O; j += 8) {                     // fc: 50 x 5520
    __builtin_prefetch((const void*)(fcw + (size_t)j * FCIN + lane), 0, 1);
    float s = 0.f;
    for (int i = lane; i < FCIN; i += 32) s += p2f[i] * fcw[(size_t)j * FCIN + i];
    s = waveSum(s);
    if (lane == 0) sfc[j] = s + fcb[j];
  }
  __syncthreads();

  if (wid == 0) {                                          // log_softmax + scatter
    float v0 = (lane < 50) ? sfc[lane] : -3.0e38f;
    float v1 = (lane + 32 < 50) ? sfc[lane + 32] : -3.0e38f;
    float m = waveMax(fmaxf(v0, v1));
    float e = ((lane < 50) ? expf(v0 - m) : 0.f) + ((lane + 32 < 50) ? expf(v1 - m) : 0.f);
    float lse = m + logf(waveSum(e));
    int base = (n < NPAR) ? (n * 100) : ((n - NPAR) * 100 + 50);
    if (lane < 50)      out[base + lane]      = v0 - lse;
    if (lane + 32 < 50) out[base + lane + 32] = v1 - lse;
  }
}

extern "C" void kernel_launch(void* const* d_in, const int* in_sizes, int n_in,
                              void* d_out, int out_size, void* d_ws, size_t ws_size,
                              hipStream_t stream) {
  const int*   quer = (const int*)d_in[0];
  const int*   doc  = (const int*)d_in[1];
  const float* emb  = (const float*)d_in[2];
  const float* w_t  = (const float*)d_in[3];
  const float* b_t  = (const float*)d_in[4];
  const float* c1w  = (const float*)d_in[5];
  const float* c1b  = (const float*)d_in[6];
  const float* c2w  = (const float*)d_in[7];
  const float* c2b  = (const float*)d_in[8];
  const float* fcw  = (const float*)d_in[9];
  const float* fcb  = (const float*)d_in[10];
  (void)in_sizes; (void)n_in; (void)out_size; (void)ws_size;

  char* ws = (char*)d_ws;
  unsigned short* qnb  = (unsigned short*)(ws + 0);        // 20480 B
  float*          tx   = (float*)(ws + 20480);             // 128 B
  unsigned short* B1   = (unsigned short*)(ws + 20736);    // 4096 B
  unsigned short* B2   = (unsigned short*)(ws + 24832);    // 46080 B
  unsigned short* xbuf = (unsigned short*)(ws + 71168);    // 2048*9000*2 = 36.9 MB

  size_t smem1 = 93376;   // dn + tyl + txl + toks + qnbl
  size_t smem2 = 249840;  // sS + sB1 + sB2 + p1 + p2 + p2f + sfc
  (void)hipFuncSetAttribute((const void*)k1_match,
                            hipFuncAttributeMaxDynamicSharedMemorySize, (int)smem1);
  (void)hipFuncSetAttribute((const void*)k2_conv,
                            hipFuncAttributeMaxDynamicSharedMemorySize, (int)smem2);

  k0_prep<<<32, 256, 0, stream>>>(quer, emb, w_t, b_t, c1w, c2w, qnb, tx, B1, B2);
  k1_match<<<NPAR, 256, smem1, stream>>>(quer, doc, emb, w_t, b_t, qnb, tx, xbuf);
  k2_conv<<<NIMG, 256, smem2, stream>>>(xbuf, B1, B2, c1b, c2b, fcw, fcb, (float*)d_out);
}
